// mTAN_enc_23304492548180
// MI455X (gfx1250) — matrix operand order repaired and checked by reading, not compiled, not run on hardware
//
#include <hip/hip_runtime.h>
#include <hip/hip_bf16.h>
#include <math.h>

// ---------------- problem constants ----------------
#define B_   64
#define L_   4096
#define LQ_  128
#define D_   32
#define ET_  128
#define H_   128
#define G3_  384      // 3*H
#define NC_  6
#define SEG_ 8
#define SEGLEN_ (L_ / SEG_)   // 512 keys per attention segment

typedef __attribute__((ext_vector_type(16))) _Float16 v16h;
typedef __attribute__((ext_vector_type(8)))  float    v8f;

__device__ __forceinline__ v8f wmma_f16(v16h a, v16h b, v8f c) {
  // D = A(16x32 f16) * B(32x16 f16) + C(16x16 f32)
  return __builtin_amdgcn_wmma_f32_16x16x32_f16(false, a, false, b, (short)0, c,
                                                false, false);
}

// ---- CDNA5 async global->LDS (ASYNCcnt-tracked DMA, bypasses VGPRs) --------
// ISA 15.18.3 opcode 98: copies global memory to the LDS address held in VDST; per-lane 16B.
__device__ __forceinline__ void async_load_b128(const _Float16* gsrc, _Float16* ldst) {
  unsigned loff = (unsigned)(uintptr_t)ldst;   // low 32 bits = LDS byte address
  asm volatile("global_load_async_to_lds_b128 %0, %1, off"
               :: "v"(loff), "v"(gsrc)
               : "memory");
}
__device__ __forceinline__ void wait_async0() {
  asm volatile("s_wait_asynccnt 0x0" ::: "memory");
}

// A fragment 16(M)x32(K) from row-major [m][k] (stride ld, halfs).
// ISA layout: lanes0-15 hold K{0..7,16..23}, lanes16-31 hold K{8..15,24..31}.
__device__ __forceinline__ v16h load_A_rm(const _Float16* p, int ld) {
  int lane = threadIdx.x & 31;
  const _Float16* r = p + (lane & 15) * ld;
  int ko = (lane < 16) ? 0 : 8;
  v16h a;
#pragma unroll
  for (int i = 0; i < 8; ++i) a[i] = r[ko + i];
#pragma unroll
  for (int i = 0; i < 8; ++i) a[8 + i] = r[16 + ko + i];
  return a;
}

// B fragment 32(K)x16(N), memory row-major [k][n]: lane%16 = N col, K{0..7,16..23} / K{8..15,24..31} per half.
__device__ __forceinline__ v16h load_B_rm(const _Float16* p, int ld) {
  int lane = threadIdx.x & 31;
  const _Float16* r = p + (lane & 15) + ((lane < 16) ? 0 : 8) * ld;
  v16h b;
#pragma unroll
  for (int i = 0; i < 16; ++i) b[i] = r[((i < 8) ? i : 8 + i) * ld];
  return b;
}

// ---------------- K0: query embedding + projection (folds 1/sqrt(ek)) -------
__global__ void qp_kernel(const float* __restrict__ qt, const float* lin_w,
                          const float* lin_b, const float* __restrict__ per_w,
                          const float* __restrict__ per_b,
                          const float* __restrict__ wq, const float* __restrict__ bq,
                          _Float16* __restrict__ qp_h) {
  __shared__ float emb[ET_];
  int tid = threadIdx.x;                     // 128 threads
  const float scale = 0.08838834764831845f;  // 1/sqrt(128)
  for (int q = 0; q < LQ_; ++q) {
    float t = qt[q];
    float e = (tid == 0) ? (t * lin_w[0] + lin_b[0])
                         : sinf(t * per_w[tid - 1] + per_b[tid - 1]);
    emb[tid] = e;
    __syncthreads();
    float acc = bq[tid];
    for (int k = 0; k < ET_; ++k) acc += emb[k] * wq[k * ET_ + tid];
    qp_h[q * ET_ + tid] = (_Float16)(acc * scale);
    __syncthreads();
  }
}

// ---------------- K1: key embedding + kp = emb@wk+bk (WMMA) -----------------
// Stores kp TRANSPOSED: kpT[b][e][l]  -> coalesced b128 stores here and
// coalesced b128 loads in the attention kernel.
__global__ void kproj_kernel(const float* __restrict__ ts, const float* lin_w,
                             const float* lin_b, const float* __restrict__ per_w,
                             const float* __restrict__ per_b,
                             const float* __restrict__ wk, const float* __restrict__ bk,
                             const float* __restrict__ x,
                             _Float16* __restrict__ kpT, _Float16* __restrict__ x_h) {
  extern __shared__ _Float16 sm1[];
  _Float16* s_emb = sm1;                 // 128 x 128 (reused as s_out[e][l])
  _Float16* s_wk  = sm1 + 128 * 128;     // 128 x 128
  int tid = threadIdx.x;                 // 256
  int b  = blockIdx.x / (L_ / 128);
  int l0 = (blockIdx.x % (L_ / 128)) * 128;
  for (int idx = tid; idx < 128 * 128; idx += 256) {
    int r = idx >> 7, c = idx & 127;
    float t = ts[(size_t)b * L_ + l0 + r];
    float e = (c == 0) ? (t * lin_w[0] + lin_b[0])
                       : sinf(t * per_w[c - 1] + per_b[c - 1]);
    s_emb[idx] = (_Float16)e;
    s_wk[idx]  = (_Float16)wk[idx];
  }
  __syncthreads();
  int w = tid >> 5, lane = tid & 31;
  int m0 = w * 16;
  int colb = lane & 15, rsel = (lane >= 16) ? 8 : 0;

  v16h afr[4];
#pragma unroll
  for (int kc = 0; kc < 4; ++kc)
    afr[kc] = load_A_rm(s_emb + m0 * 128 + kc * 32, 128);
  v8f acc[8];
#pragma unroll
  for (int nt = 0; nt < 8; ++nt) acc[nt] = (v8f){};
#pragma unroll
  for (int kc = 0; kc < 4; ++kc) {
#pragma unroll
    for (int nt = 0; nt < 8; ++nt) {
      v16h bb = load_B_rm(s_wk + (kc * 32) * 128 + nt * 16, 128);
      acc[nt] = wmma_f16(afr[kc], bb, acc[nt]);
    }
  }
  __syncthreads();   // everyone is done reading s_emb -> reuse as s_out[e][l]
#pragma unroll
  for (int nt = 0; nt < 8; ++nt) {
    int col = nt * 16 + colb;
    float bkv = bk[col];
#pragma unroll
    for (int r = 0; r < 8; ++r) {
      int row = m0 + r + rsel;
      s_emb[col * 128 + row] = (_Float16)(acc[nt][r] + bkv);
    }
  }
  __syncthreads();
  // coalesced transposed store: kpT[b][e][l0..l0+128]
  for (int idx = tid; idx < 128 * 16; idx += 256) {
    int e = idx >> 4, ch = idx & 15;  // 16 x uint4 (8 halfs) per e-row
    *(uint4*)(kpT + ((size_t)b * ET_ + e) * L_ + l0 + ch * 8) =
        *(const uint4*)(s_emb + e * 128 + ch * 8);
  }
  // x -> f16
  for (int idx = tid; idx < 128 * D_; idx += 256) {
    int r = idx >> 5, c = idx & 31;
    size_t g = ((size_t)b * L_ + l0 + r) * D_ + c;
    x_h[g] = (_Float16)x[g];
  }
}

// ---------------- K2: flash-attention partials (split-K over L) -------------
// grid (B, SEG), 256 thr = 8 waves; wave w owns query strip [16w,16w+16).
// Double-buffered async global->LDS copies overlap the next tile's DMA with
// the current tile's WMMA + online softmax.
__global__ void attn_kernel(const _Float16* __restrict__ qp_h,
                            const _Float16* __restrict__ kpT,
                            const _Float16* __restrict__ x_h,
                            float* __restrict__ part_att,
                            float* __restrict__ part_m,
                            float* __restrict__ part_s) {
  __shared__ __align__(16) _Float16 s_kpT[2][128 * 32];  // [buf][e][key]
  __shared__ __align__(16) _Float16 s_x[2][32 * 32];     // [buf][key][d]
  __shared__ __align__(16) _Float16 s_p[8 * 16 * 32];    // per-wave P stage
  int tid = threadIdx.x, w = tid >> 5, lane = tid & 31;
  int b = blockIdx.x, seg = blockIdx.y;
  int k_base = seg * SEGLEN_;
  int colb = lane & 15;
  int rsel = (lane >= 16) ? 8 : 0;

  v16h aq[4];
#pragma unroll
  for (int kc = 0; kc < 4; ++kc)
    aq[kc] = load_A_rm(qp_h + (w * 16) * ET_ + kc * 32, ET_);

  v8f acc0 = {}, acc1 = {};
  float rm[8], rs[8];
#pragma unroll
  for (int r = 0; r < 8; ++r) { rm[r] = -1e30f; rs[r] = 0.f; }

  const _Float16* kp_b = kpT + (size_t)b * ET_ * L_;
  const _Float16* x_b  = x_h + (size_t)b * L_ * D_;
  _Float16* sp = s_p + w * 16 * 32;

  auto issue_tile = [&](int kt, int buf) {
    int k0 = k_base + kt * 32;
    // kp tile: 128 e-rows x 32 keys (8 KB), 2 async b128 per thread
    for (int idx = tid; idx < 128 * 4; idx += 256) {
      int e = idx >> 2, ch = idx & 3;
      async_load_b128(kp_b + (size_t)e * L_ + k0 + ch * 8,
                      &s_kpT[buf][e * 32 + ch * 8]);
    }
    // x tile: 32 keys x 32 dims (2 KB)
    if (tid < 128) {
      int key = tid >> 2, ch = tid & 3;
      async_load_b128(x_b + (size_t)(k0 + key) * D_ + ch * 8,
                      &s_x[buf][key * 32 + ch * 8]);
    }
  };

  issue_tile(0, 0);
  const int NT = SEGLEN_ / 32;
  for (int kt = 0; kt < NT; ++kt) {
    int buf = kt & 1;
    wait_async0();      // my DMA for tile kt complete
    __syncthreads();    // everyone's DMA complete; buf^1 free for reuse
    if (kt + 1 < NT) issue_tile(kt + 1, buf ^ 1);

    // scores for 32 keys: two 16x16 C tiles (B frags contiguous in LDS)
    v8f c0 = {}, c1 = {};
#pragma unroll
    for (int kc = 0; kc < 4; ++kc) {
      v16h b0 = load_B_rm(&s_kpT[buf][(kc * 32) * 32 + 0], 32);
      v16h b1 = load_B_rm(&s_kpT[buf][(kc * 32) * 32 + 16], 32);
      c0 = wmma_f16(aq[kc], b0, c0);
      c1 = wmma_f16(aq[kc], b1, c1);
    }

    // online softmax (row reductions across the 16-lane half-waves)
#pragma unroll
    for (int r = 0; r < 8; ++r) {
      float t = fmaxf(c0[r], c1[r]);
#pragma unroll
      for (int off = 8; off >= 1; off >>= 1) t = fmaxf(t, __shfl_xor(t, off, 32));
      float nm = fmaxf(rm[r], t);
      float corr = __expf(rm[r] - nm);
      rm[r] = nm;
      float e0 = __expf(c0[r] - nm);
      float e1 = __expf(c1[r] - nm);
      float sum = e0 + e1;
#pragma unroll
      for (int off = 8; off >= 1; off >>= 1) sum += __shfl_xor(sum, off, 32);
      rs[r] = rs[r] * corr + sum;
      acc0[r] *= corr; acc1[r] *= corr;
      sp[(r + rsel) * 32 + colb]      = (_Float16)e0;   // C-layout -> row-major stage
      sp[(r + rsel) * 32 + 16 + colb] = (_Float16)e1;
    }
    __syncthreads();

    // att += P(16x32) @ X(32keys x 32dims)
    v16h ap  = load_A_rm(sp, 32);
    v16h bx0 = load_B_rm(&s_x[buf][0], 32);
    v16h bx1 = load_B_rm(&s_x[buf][16], 32);
    acc0 = wmma_f16(ap, bx0, acc0);
    acc1 = wmma_f16(ap, bx1, acc1);
  }

  size_t pbase = ((size_t)b * SEG_ + seg) * LQ_;
#pragma unroll
  for (int r = 0; r < 8; ++r) {
    int q = w * 16 + r + rsel;
    part_att[(pbase + q) * D_ + colb]      = acc0[r];
    part_att[(pbase + q) * D_ + 16 + colb] = acc1[r];
    if (colb == 0) { part_m[pbase + q] = rm[r]; part_s[pbase + q] = rs[r]; }
  }
}

// ---------------- K3: combine segments + att_out = att@wo + bo --------------
__global__ void combine_kernel(const float* __restrict__ part_att,
                               const float* __restrict__ part_m,
                               const float* __restrict__ part_s,
                               const float* __restrict__ wo, const float* __restrict__ bo,
                               _Float16* __restrict__ xs0) {
  int b = blockIdx.x, q = threadIdx.x;  // 128 threads
  float M = -1e30f;
  for (int s = 0; s < SEG_; ++s)
    M = fmaxf(M, part_m[(size_t)(b * SEG_ + s) * LQ_ + q]);
  float att[D_];
  for (int d = 0; d < D_; ++d) att[d] = 0.f;
  float denom = 0.f;
  for (int s = 0; s < SEG_; ++s) {
    size_t idx = (size_t)(b * SEG_ + s) * LQ_ + q;
    float wgt = __expf(part_m[idx] - M);
    denom += wgt * part_s[idx];
    const float* pa = part_att + idx * D_;
    for (int d = 0; d < D_; ++d) att[d] += wgt * pa[d];
  }
  float inv = 1.0f / denom;
  for (int d = 0; d < D_; ++d) att[d] *= inv;
  for (int o = 0; o < H_; ++o) {
    float a = bo[o];
    for (int d = 0; d < D_; ++d) a += att[d] * wo[d * H_ + o];
    xs0[((size_t)b * LQ_ + q) * H_ + o] = (_Float16)a;
  }
}

// ------- weight convert + transpose: src [dir][N][K] f32 -> dst [dir][K][N] f16
__global__ void wconvT(const float* __restrict__ src, _Float16* __restrict__ dst,
                       int N, int K) {
  int dir = blockIdx.y;
  int i = blockIdx.x * 256 + threadIdx.x;
  if (i >= N * K) return;
  int n = i / K, k = i % K;
  dst[((size_t)dir * K + k) * N + n] = (_Float16)src[(size_t)dir * N * K + i];
}

// ---------------- K4: gi = xs @ wih^T + bih (WMMA GEMM, both dirs) ----------
// wihT is pre-transposed [dir][K][G3] f16 -> B frags are read row-major [k][n] (ld = G3).
template <int KD>
__global__ void gi_gemm(const _Float16* __restrict__ xs,
                        const _Float16* __restrict__ wihT,
                        const float* __restrict__ bih, float* __restrict__ gi) {
  int dir = blockIdx.y;
  int w = threadIdx.x >> 5, lane = threadIdx.x & 31;
  int m0 = blockIdx.x * 128 + w * 16;                 // M = B*LQ = 8192
  const _Float16* wTd = wihT + (size_t)dir * KD * G3_;
  const float* bd = bih + dir * G3_;
  float* gid = gi + (size_t)dir * (B_ * LQ_) * G3_;
  v16h a[KD / 32];
#pragma unroll
  for (int kc = 0; kc < KD / 32; ++kc)
    a[kc] = load_A_rm(xs + (size_t)m0 * KD + kc * 32, KD);
  for (int nt = 0; nt < G3_ / 16; ++nt) {
    v8f acc = {};
#pragma unroll
    for (int kc = 0; kc < KD / 32; ++kc) {
      v16h bb = load_B_rm(wTd + (size_t)(kc * 32) * G3_ + nt * 16, G3_);
      acc = wmma_f16(a[kc], bb, acc);
    }
    int col = nt * 16 + (lane & 15);
    float bv = bd[col];
#pragma unroll
    for (int r = 0; r < 8; ++r) {
      int row = m0 + r + ((lane >= 16) ? 8 : 0);
      gid[(size_t)row * G3_ + col] = acc[r] + bv;
    }
  }
}

// ---------------- K5: GRU recurrent scan (one block per direction) ----------
// whhT ([dir][128][384] f16) async-DMA'd into LDS once; per step gh = h @ whh^T.
__global__ void gru_scan(const float* __restrict__ gi,
                         const _Float16* __restrict__ whhT,
                         const float* __restrict__ bhh,
                         _Float16* __restrict__ ys,       // layer0: (B,LQ,2H) f16; else null
                         float* __restrict__ class_out) { // layer1: (B,2H) f32; else null
  extern __shared__ char smraw[];
  float*    s_gh = (float*)smraw;                                   // 64 x 384
  float*    s_h  = (float*)(smraw + B_ * G3_ * 4);                  // 64 x 128
  _Float16* s_hh = (_Float16*)(smraw + B_ * G3_ * 4 + B_ * H_ * 4); // 64 x 128
  _Float16* s_w  = (_Float16*)(smraw + B_ * G3_ * 4 + B_ * H_ * 4 + B_ * H_ * 2); // 128x384
  int dir = blockIdx.x;
  int tid = threadIdx.x, w = tid >> 5, lane = tid & 31;
  const _Float16* wTd = whhT + (size_t)dir * H_ * G3_;
  const float*    bd  = bhh + dir * G3_;
  const float*    gid = gi + (size_t)dir * (B_ * LQ_) * G3_;
  for (int i = tid; i < B_ * H_; i += 256) { s_h[i] = 0.f; s_hh[i] = (_Float16)0.f; }
  for (int idx = tid; idx < H_ * G3_ / 8; idx += 256)      // 96 KB weight DMA
    async_load_b128(wTd + idx * 8, s_w + idx * 8);
  wait_async0();
  __syncthreads();
  for (int step = 0; step < LQ_; ++step) {
    int t = dir ? (LQ_ - 1 - step) : step;
    // gh tiles: 4 Mtiles x 24 Ntiles = 96, round-robin over 8 waves
    for (int tile = w; tile < 96; tile += 8) {
      int mt = tile & 3, nt = tile >> 2;
      v8f acc = {};
#pragma unroll
      for (int kc = 0; kc < 4; ++kc) {
        v16h a  = load_A_rm(s_hh + (mt * 16) * H_ + kc * 32, H_);
        v16h bb = load_B_rm(s_w + (kc * 32) * G3_ + nt * 16, G3_);
        acc = wmma_f16(a, bb, acc);
      }
      int col = nt * 16 + (lane & 15);
      float bv = bd[col];
#pragma unroll
      for (int r = 0; r < 8; ++r) {
        int row = mt * 16 + r + ((lane >= 16) ? 8 : 0);
        s_gh[row * G3_ + col] = acc[r] + bv;
      }
    }
    __syncthreads();
    for (int i = tid; i < B_ * H_; i += 256) {
      int bb = i >> 7, j = i & 127;
      const float* girow = gid + ((size_t)bb * LQ_ + t) * G3_;
      float ir = girow[j], iz = girow[H_ + j], inn = girow[2 * H_ + j];
      float hr = s_gh[bb * G3_ + j];
      float hz = s_gh[bb * G3_ + H_ + j];
      float hn = s_gh[bb * G3_ + 2 * H_ + j];
      float r = 1.f / (1.f + __expf(-(ir + hr)));
      float z = 1.f / (1.f + __expf(-(iz + hz)));
      float n = tanhf(inn + r * hn);
      float h = (1.f - z) * n + z * s_h[i];
      s_h[i] = h; s_hh[i] = (_Float16)h;
      if (ys) ys[((size_t)bb * LQ_ + t) * (2 * H_) + dir * H_ + j] = (_Float16)h;
    }
    __syncthreads();
  }
  if (class_out) {
    // fwd final h == g1[:, -1, :H]; bwd final h (scan ends at t=0) == g1[:, 0, H:]
    for (int i = tid; i < B_ * H_; i += 256) {
      int bb = i >> 7, j = i & 127;
      class_out[bb * (2 * H_) + dir * H_ + j] = s_h[i];
    }
  }
}

// ---------------- K6: classifier head ---------------------------------------
__global__ void classifier_kernel(const float* __restrict__ cin,
                                  const float* __restrict__ c1w, const float* __restrict__ c1b,
                                  const float* __restrict__ c2w, const float* __restrict__ c2b,
                                  const float* __restrict__ c3w, const float* __restrict__ c3b,
                                  float* __restrict__ out) {
  int b = threadIdx.x;  // 64 threads
  if (b >= B_) return;
  float h1[128];
  for (int o = 0; o < 128; ++o) {
    float a = c1b[o];
    for (int k = 0; k < 256; ++k) a += cin[b * 256 + k] * c1w[k * 128 + o];
    h1[o] = a;
  }
  float h2[64];
  for (int o = 0; o < 64; ++o) {
    float a = c2b[o];
    for (int k = 0; k < 128; ++k) a += h1[k] * c2w[k * 64 + o];
    h2[o] = a;
  }
  for (int o = 0; o < NC_; ++o) {
    float a = c3b[o];
    for (int k = 0; k < 64; ++k) a += h2[k] * c3w[k * NC_ + o];
    out[b * NC_ + o] = a;
  }
}

// ---------------- host side --------------------------------------------------
extern "C" void kernel_launch(void* const* d_in, const int* in_sizes, int n_in,
                              void* d_out, int out_size, void* d_ws, size_t ws_size,
                              hipStream_t stream) {
  const float* x      = (const float*)d_in[0];
  const float* ts     = (const float*)d_in[1];
  const float* qt     = (const float*)d_in[2];
  const float* lin_w  = (const float*)d_in[3];
  const float* lin_b  = (const float*)d_in[4];
  const float* per_w  = (const float*)d_in[5];
  const float* per_b  = (const float*)d_in[6];
  const float* wq     = (const float*)d_in[7];
  const float* bq     = (const float*)d_in[8];
  const float* wk     = (const float*)d_in[9];
  const float* bk     = (const float*)d_in[10];
  const float* wo     = (const float*)d_in[11];
  const float* bo     = (const float*)d_in[12];
  const float* g0_wih = (const float*)d_in[13];
  const float* g0_whh = (const float*)d_in[14];
  const float* g0_bih = (const float*)d_in[15];
  const float* g0_bhh = (const float*)d_in[16];
  const float* g1_wih = (const float*)d_in[17];
  const float* g1_whh = (const float*)d_in[18];
  const float* g1_bih = (const float*)d_in[19];
  const float* g1_bhh = (const float*)d_in[20];
  const float* c1w = (const float*)d_in[21];
  const float* c1b = (const float*)d_in[22];
  const float* c2w = (const float*)d_in[23];
  const float* c2b = (const float*)d_in[24];
  const float* c3w = (const float*)d_in[25];
  const float* c3b = (const float*)d_in[26];
  float* out = (float*)d_out;

  char* ws = (char*)d_ws;
  size_t off = 0;
  auto take = [&](size_t bytes) -> char* {
    char* p = ws + off;
    off += (bytes + 255) & ~(size_t)255;
    return p;
  };
  _Float16* qp_h     = (_Float16*)take((size_t)LQ_ * ET_ * 2);
  _Float16* kpT      = (_Float16*)take((size_t)B_ * ET_ * L_ * 2);      // 64 MB, [b][e][l]
  _Float16* x_h      = (_Float16*)take((size_t)B_ * L_ * D_ * 2);       // 16 MB
  float*    part_att = (float*)take((size_t)B_ * SEG_ * LQ_ * D_ * 4);  // 8 MB
  float*    part_m   = (float*)take((size_t)B_ * SEG_ * LQ_ * 4);
  float*    part_s   = (float*)take((size_t)B_ * SEG_ * LQ_ * 4);
  _Float16* xs0_h    = (_Float16*)take((size_t)B_ * LQ_ * H_ * 2);      // GRU0 input
  _Float16* g0seq_h  = (_Float16*)take((size_t)B_ * LQ_ * 2 * H_ * 2);  // GRU1 input
  float*    gi_buf   = (float*)take((size_t)2 * B_ * LQ_ * G3_ * 4);    // 24 MB
  float*    class_in = (float*)take((size_t)B_ * 2 * H_ * 4);
  _Float16* wih0T    = (_Float16*)take((size_t)2 * H_ * G3_ * 2);       // [dir][128][384]
  _Float16* wih1T    = (_Float16*)take((size_t)2 * 2 * H_ * G3_ * 2);   // [dir][256][384]
  _Float16* whh0T    = (_Float16*)take((size_t)2 * H_ * G3_ * 2);
  _Float16* whh1T    = (_Float16*)take((size_t)2 * H_ * G3_ * 2);
  (void)in_sizes; (void)n_in; (void)out_size; (void)ws_size;

  // 1. query projection
  qp_kernel<<<1, 128, 0, stream>>>(qt, lin_w, lin_b, per_w, per_b, wq, bq, qp_h);
  // 2. key embedding + projection (WMMA), kp stored transposed; x -> f16
  kproj_kernel<<<B_ * (L_ / 128), 256, 2 * 128 * 128 * 2, stream>>>(
      ts, lin_w, lin_b, per_w, per_b, wk, bk, x, kpT, x_h);
  // 3. flash attention partials (WMMA + async double-buffered tiles)
  attn_kernel<<<dim3(B_, SEG_), 256, 0, stream>>>(qp_h, kpT, x_h,
                                                  part_att, part_m, part_s);
  // 4. combine + output projection -> GRU0 input (f16)
  combine_kernel<<<B_, 128, 0, stream>>>(part_att, part_m, part_s, wo, bo, xs0_h);
  // 5. GRU weights -> f16, transposed to [dir][K][3H]
  wconvT<<<dim3((G3_ * H_ + 255) / 256, 2), 256, 0, stream>>>(g0_wih, wih0T, G3_, H_);
  wconvT<<<dim3((G3_ * 2 * H_ + 255) / 256, 2), 256, 0, stream>>>(g1_wih, wih1T, G3_, 2 * H_);
  wconvT<<<dim3((G3_ * H_ + 255) / 256, 2), 256, 0, stream>>>(g0_whh, whh0T, G3_, H_);
  wconvT<<<dim3((G3_ * H_ + 255) / 256, 2), 256, 0, stream>>>(g1_whh, whh1T, G3_, H_);
  // 6. layer-0: gi GEMM (WMMA) + scan (WMMA per step, weights in LDS)
  size_t scan_lds = (size_t)B_ * G3_ * 4 + (size_t)B_ * H_ * 4 + (size_t)B_ * H_ * 2 +
                    (size_t)H_ * G3_ * 2;   // 240 KB < 320 KB/WGP
  gi_gemm<128><<<dim3(B_ * LQ_ / 128, 2), 256, 0, stream>>>(xs0_h, wih0T, g0_bih, gi_buf);
  gru_scan<<<2, 256, scan_lds, stream>>>(gi_buf, whh0T, g0_bhh, g0seq_h, nullptr);
  // 7. layer-1
  gi_gemm<256><<<dim3(B_ * LQ_ / 128, 2), 256, 0, stream>>>(g0seq_h, wih1T, g1_bih, gi_buf);
  gru_scan<<<2, 256, scan_lds, stream>>>(gi_buf, whh1T, g1_bhh, nullptr, class_in);
  // 8. classifier head
  classifier_kernel<<<1, 64, 0, stream>>>(class_in, c1w, c1b, c2w, c2b, c3w, c3b, out);
}
